// DeepCrossNetworkModel_Controller_hard_5677946765435
// MI455X (gfx1250) — compile-verified, matrix operand order checked by reading, not run on hardware
//
#include <hip/hip_runtime.h>
#include <hip/hip_bf16.h>
#include <stdint.h>

namespace {

constexpr int BB    = 16384;
constexpr int FF    = 39;
constexpr int EE    = 64;
constexpr int LL    = 6;
constexpr int KTOP  = 16;
constexpr int VOCAB = 10000;
constexpr int DD    = FF * EE;   // 2496
constexpr int H0    = 1024;
constexpr int H1    = 512;
constexpr float EPS = 1e-5f;

typedef __attribute__((ext_vector_type(16))) __bf16 v16bf;
typedef __attribute__((ext_vector_type(8)))  float  v8f;

union Frag { v16bf v; uint4 u[2]; };

__device__ __forceinline__ float wave_sum(float v) {
  #pragma unroll
  for (int o = 16; o; o >>= 1) v += __shfl_xor(v, o, 32);
  return v;
}
__device__ __forceinline__ float wave_max(float v) {
  #pragma unroll
  for (int o = 16; o; o >>= 1) v = fmaxf(v, __shfl_xor(v, o, 32));
  return v;
}

// ---------------- parameter prep ----------------
__global__ void prep_params_kernel(
    const float* g, const float* be, const float* m, const float* v,
    const float* b0, const float* g0, const float* be0, const float* m0, const float* v0,
    const float* b1, const float* g1, const float* be1, const float* m1, const float* v1,
    float* bnS, float* bnB, float* c0S, float* c0B, float* c1S, float* c1B)
{
  int i = threadIdx.x;
  if (i < FF) {
    float s = g[i] * rsqrtf(v[i] + EPS);
    bnS[i] = s;
    bnB[i] = be[i] - m[i] * s;
  }
  if (i < H0) {
    float s = g0[i] * rsqrtf(v0[i] + EPS);
    c0S[i] = s;
    c0B[i] = be0[i] + (b0[i] - m0[i]) * s;
  }
  if (i < H1) {
    float s = g1[i] * rsqrtf(v1[i] + EPS);
    c1S[i] = s;
    c1B[i] = be1[i] + (b1[i] - m1[i]) * s;
  }
}

// in: K x N row-major f32 ; out: N x K row-major bf16 (weight pre-transpose)
__global__ void transpose_to_bf16_kernel(const float* __restrict__ in, __bf16* __restrict__ out,
                                         int K, int N) {
  int i = blockIdx.x * blockDim.x + threadIdx.x;   // over N*K (exact multiple of 256)
  int n = i / K, k = i - n * K;
  out[i] = (__bf16)in[(size_t)k * N + n];
}

// ctrl_w (DD x 39, row-major) -> padded transposed bf16 (48 x DD)
__global__ void build_ctrlT_kernel(const float* __restrict__ ctrl_w, __bf16* __restrict__ ctrlT) {
  int i = blockIdx.x * blockDim.x + threadIdx.x;   // over 48*DD
  int n = i / DD, k = i - n * DD;
  float v = (n < FF) ? ctrl_w[k * FF + n] : 0.f;
  ctrlT[i] = (__bf16)v;
}

// ---------------- embedding gather + feature BN ----------------
__global__ void gather_bn_kernel(const int* __restrict__ x, const float* __restrict__ table,
                                 const float* __restrict__ bnS, const float* __restrict__ bnB,
                                 float* __restrict__ emb_f32, __bf16* __restrict__ flat_bf16)
{
  int g = blockIdx.x * blockDim.x + threadIdx.x;   // over BB*DD (exact)
  int b = g / DD, r = g - b * DD;
  int f = r >> 6, e = r & 63;
  int tok = x[b * FF + f] + f * VOCAB;
  float v = table[(size_t)tok * EE + e];
  v = v * bnS[f] + bnB[f];
  emb_f32[g] = v;
  flat_bf16[g] = (__bf16)v;
}

// ---------------- controller GEMM: logits = flat @ ctrl_w + b  (N padded to 48) ----------------
__global__ __launch_bounds__(32)
void ctrl_gemm_kernel(const __bf16* __restrict__ A, const __bf16* __restrict__ Bt,
                      const float* __restrict__ ctrl_b, float* __restrict__ logits)
{
  int lane = threadIdx.x;
  int l16 = lane & 15, hf = lane >> 4;
  int rowBase = blockIdx.x * 16;
  v8f acc[3] = {};
  const __bf16* aRow = A + (size_t)(rowBase + l16) * DD + hf * 8;
  for (int kt = 0; kt < DD / 32; ++kt) {
    Frag a;
    a.u[0] = *(const uint4*)(aRow + kt * 32);
    a.u[1] = *(const uint4*)(aRow + kt * 32 + 16);
    #pragma unroll
    for (int ni = 0; ni < 3; ++ni) {
      Frag b;
      const __bf16* p = Bt + (size_t)(ni * 16 + l16) * DD + kt * 32 + hf * 8;
      b.u[0] = *(const uint4*)p;
      b.u[1] = *(const uint4*)(p + 16);
      acc[ni] = __builtin_amdgcn_wmma_f32_16x16x32_bf16(false, a.v, false, b.v,
                                                        (short)0, acc[ni], false, false);
    }
  }
  #pragma unroll
  for (int ni = 0; ni < 3; ++ni) {
    int col = ni * 16 + l16;
    float bias = (col < FF) ? ctrl_b[col] : 0.f;
    #pragma unroll
    for (int r = 0; r < 8; ++r) {
      int row = rowBase + r + hf * 8;
      logits[(size_t)row * 48 + col] = acc[ni][r] + bias;
    }
  }
}

// ---------------- softmax + top-16 + normalize + mask-apply (wave per row) ----------------
__global__ __launch_bounds__(256)
void topk_mask_kernel(const float* __restrict__ logits, float* __restrict__ emb_f32,
                      __bf16* __restrict__ flat_bf16)
{
  __shared__ float maskS[8][FF];
  int lane = threadIdx.x & 31;
  int w = threadIdx.x >> 5;
  int row = blockIdx.x * 8 + w;
  const float NEG = -1e30f;
  float l1 = (lane < FF)      ? logits[(size_t)row * 48 + lane]      : NEG;
  float l2 = (lane + 32 < FF) ? logits[(size_t)row * 48 + lane + 32] : NEG;
  float mx = wave_max(fmaxf(l1, l2));
  float e1 = (lane < FF)      ? __expf(l1 - mx) : 0.f;
  float e2 = (lane + 32 < FF) ? __expf(l2 - mx) : 0.f;
  float inv = 1.f / wave_sum(e1 + e2);
  float w1 = e1 * inv, w2 = e2 * inv;
  float a1 = (lane < FF)      ? w1 : -1.f;
  float a2 = (lane + 32 < FF) ? w2 : -1.f;
  float m1 = 0.f, m2 = 0.f, tsum = 0.f;
  for (int it = 0; it < KTOP; ++it) {
    float bv; int bi;
    if (a1 >= a2) { bv = a1; bi = lane; } else { bv = a2; bi = lane + 32; }
    #pragma unroll
    for (int o = 16; o; o >>= 1) {
      float ov = __shfl_xor(bv, o, 32);
      int   oi = __shfl_xor(bi, o, 32);
      if (ov > bv || (ov == bv && oi < bi)) { bv = ov; bi = oi; }
    }
    tsum += bv;
    if (bi == lane)      { m1 = w1; a1 = -1.f; }
    if (bi == lane + 32) { m2 = w2; a2 = -1.f; }
  }
  float nf = 1.f / tsum;
  if (lane < FF)      maskS[w][lane]      = m1 * nf;
  if (lane + 32 < FF) maskS[w][lane + 32] = m2 * nf;
  __syncthreads();
  size_t base = (size_t)row * DD;
  for (int t = 0; t < DD / 32; ++t) {
    int j = lane + t * 32;
    float mk = maskS[w][j >> 6];
    float v = emb_f32[base + j] * mk;
    emb_f32[base + j] = v;
    flat_bf16[base + j] = (__bf16)v;
  }
}

// ---------------- tiled bf16 WMMA GEMM, double-buffered LDS, fused BN+ReLU ----------------
// C[M,N] = relu((A[M,K] @ W[K,N]) * cS[n] + cB[n]), with W supplied pre-transposed WT[N,K].
// M%128==N%128==K%32==0.
template<typename OutT>
__global__ __launch_bounds__(256)
void gemm_bn_relu_kernel(const __bf16* __restrict__ A, const __bf16* __restrict__ WT,
                         const float* __restrict__ cS, const float* __restrict__ cB,
                         OutT* __restrict__ out, int N, int K)
{
  constexpr int LDT = 40;                 // padded bf16 stride (keeps 16B alignment)
  __shared__ __bf16 As[2][128 * LDT];     // A tile 128x32 (row-major), ping-pong
  __shared__ __bf16 Bs[2][128 * LDT];     // B tile as Bs[n][k], ping-pong
  const int tid = threadIdx.x;
  const int lane = tid & 31, wid = tid >> 5;
  const int waveM = wid & 3, waveN = wid >> 2;   // 4x2 wave grid -> wave tile 32x64
  const int bm = blockIdx.y * 128, bn = blockIdx.x * 128;
  const int l16 = lane & 15, hf = lane >> 4;

  // each thread copies two 16B chunks of A and two of B per k-step
  const int rr = tid >> 2, kk = (tid & 3) * 8;
  const __bf16* gA0 = A  + (size_t)(bm + rr) * K + kk;
  const __bf16* gA1 = A  + (size_t)(bm + rr + 64) * K + kk;
  const __bf16* gB0 = WT + (size_t)(bn + rr) * K + kk;
  const __bf16* gB1 = WT + (size_t)(bn + rr + 64) * K + kk;

  v8f acc[2][4] = {};

  // prologue: stage tile 0
  *(uint4*)(&As[0][rr * LDT + kk])        = *(const uint4*)gA0;
  *(uint4*)(&As[0][(rr + 64) * LDT + kk]) = *(const uint4*)gA1;
  *(uint4*)(&Bs[0][rr * LDT + kk])        = *(const uint4*)gB0;
  *(uint4*)(&Bs[0][(rr + 64) * LDT + kk]) = *(const uint4*)gB1;
  __syncthreads();

  const int nk = K / 32;
  for (int kt = 0; kt < nk; ++kt) {
    const int cur = kt & 1, nxt = cur ^ 1;
    const bool hasNext = (kt + 1 < nk);
    uint4 ta0, ta1, tb0, tb1;
    if (hasNext) {                      // in-flight during the WMMA block below
      const int ko = (kt + 1) * 32;
      ta0 = *(const uint4*)(gA0 + ko);
      ta1 = *(const uint4*)(gA1 + ko);
      tb0 = *(const uint4*)(gB0 + ko);
      tb1 = *(const uint4*)(gB1 + ko);
    }
    // fragment loads: lane holds row/col (lane&15), K chunks hf*8 and 16+hf*8
    Frag af[2], bfr[4];
    #pragma unroll
    for (int mi = 0; mi < 2; ++mi) {
      const __bf16* p = &As[cur][(waveM * 32 + mi * 16 + l16) * LDT + hf * 8];
      af[mi].u[0] = *(const uint4*)p;
      af[mi].u[1] = *(const uint4*)(p + 16);
    }
    #pragma unroll
    for (int ni = 0; ni < 4; ++ni) {
      const __bf16* p = &Bs[cur][(waveN * 64 + ni * 16 + l16) * LDT + hf * 8];
      bfr[ni].u[0] = *(const uint4*)p;
      bfr[ni].u[1] = *(const uint4*)(p + 16);
    }
    #pragma unroll
    for (int mi = 0; mi < 2; ++mi)
      #pragma unroll
      for (int ni = 0; ni < 4; ++ni)
        acc[mi][ni] = __builtin_amdgcn_wmma_f32_16x16x32_bf16(false, af[mi].v, false, bfr[ni].v,
                                                              (short)0, acc[mi][ni], false, false);
    if (hasNext) {
      *(uint4*)(&As[nxt][rr * LDT + kk])        = ta0;
      *(uint4*)(&As[nxt][(rr + 64) * LDT + kk]) = ta1;
      *(uint4*)(&Bs[nxt][rr * LDT + kk])        = tb0;
      *(uint4*)(&Bs[nxt][(rr + 64) * LDT + kk]) = tb1;
    }
    __syncthreads();   // one barrier per k-step (covers reads of cur + stores to nxt)
  }
  // epilogue: per-column BN scale/bias + ReLU
  #pragma unroll
  for (int ni = 0; ni < 4; ++ni) {
    int col = bn + waveN * 64 + ni * 16 + l16;
    float s = cS[col], b = cB[col];
    #pragma unroll
    for (int mi = 0; mi < 2; ++mi) {
      int rbase = bm + waveM * 32 + mi * 16 + hf * 8;
      #pragma unroll
      for (int r = 0; r < 8; ++r) {
        float v = fmaxf(acc[mi][ni][r] * s + b, 0.f);
        out[(size_t)(rbase + r) * N + col] = (OutT)v;
      }
    }
  }
}

// ---------------- cross network (wave per row, xl/x0 register-resident) ----------------
__global__ __launch_bounds__(256)
void cross_kernel(const float* __restrict__ x0buf, const float* __restrict__ cw,
                  const float* __restrict__ cb, const float* __restrict__ lin_w,
                  const float* __restrict__ lin_b, float* __restrict__ pp)
{
  __shared__ float cwS[LL * DD];   // 58.5 KB
  for (int i = threadIdx.x; i < LL * DD; i += 256) cwS[i] = cw[i];
  __syncthreads();
  int lane = threadIdx.x & 31, w = threadIdx.x >> 5;
  int row = blockIdx.x * 8 + w;
  size_t base = (size_t)row * DD;
  float x0[DD / 32], xl[DD / 32];
  #pragma unroll
  for (int t = 0; t < DD / 32; ++t) { x0[t] = x0buf[base + lane + t * 32]; xl[t] = x0[t]; }
  for (int l = 0; l < LL; ++l) {
    float d = 0.f;
    #pragma unroll
    for (int t = 0; t < DD / 32; ++t) d += xl[t] * cwS[l * DD + lane + t * 32];
    d = wave_sum(d);
    const float* cbl = cb + (size_t)l * DD;
    #pragma unroll
    for (int t = 0; t < DD / 32; ++t) xl[t] += x0[t] * d + cbl[lane + t * 32];
  }
  float acc = 0.f;
  #pragma unroll
  for (int t = 0; t < DD / 32; ++t) acc += xl[t] * lin_w[lane + t * 32];
  acc = wave_sum(acc);
  if (lane == 0) pp[row] = acc + lin_b[0];
}

// ---------------- final dot over h1 + sigmoid ----------------
__global__ __launch_bounds__(256)
void final_kernel(const float* __restrict__ h1, const float* __restrict__ lin_w,
                  const float* __restrict__ pp, float* __restrict__ out)
{
  int lane = threadIdx.x & 31, w = threadIdx.x >> 5;
  int row = blockIdx.x * 8 + w;
  float s = 0.f;
  #pragma unroll
  for (int t = 0; t < H1 / 32; ++t)
    s += h1[(size_t)row * H1 + lane + t * 32] * lin_w[DD + lane + t * 32];
  s = wave_sum(s);
  if (lane == 0) {
    float p = pp[row] + s;
    out[row] = 1.f / (1.f + __expf(-p));
  }
}

} // namespace

extern "C" void kernel_launch(void* const* d_in, const int* in_sizes, int n_in,
                              void* d_out, int out_size, void* d_ws, size_t ws_size,
                              hipStream_t stream)
{
  (void)in_sizes; (void)n_in; (void)out_size; (void)ws_size;
  const int*   x         = (const int*)  d_in[0];
  const float* emb_table = (const float*)d_in[1];
  const float* bn_gamma  = (const float*)d_in[2];
  const float* bn_beta   = (const float*)d_in[3];
  const float* bn_mean   = (const float*)d_in[4];
  const float* bn_var    = (const float*)d_in[5];
  const float* ctrl_w    = (const float*)d_in[6];
  const float* ctrl_b    = (const float*)d_in[7];
  const float* cross_w   = (const float*)d_in[8];
  const float* cross_b   = (const float*)d_in[9];
  const float* mlp_w0    = (const float*)d_in[10];
  const float* mlp_b0    = (const float*)d_in[11];
  const float* mlp_g0    = (const float*)d_in[12];
  const float* mlp_be0   = (const float*)d_in[13];
  const float* mlp_m0    = (const float*)d_in[14];
  const float* mlp_v0    = (const float*)d_in[15];
  const float* mlp_w1    = (const float*)d_in[16];
  const float* mlp_b1    = (const float*)d_in[17];
  const float* mlp_g1    = (const float*)d_in[18];
  const float* mlp_be1   = (const float*)d_in[19];
  const float* mlp_m1    = (const float*)d_in[20];
  const float* mlp_v1    = (const float*)d_in[21];
  const float* lin_w     = (const float*)d_in[22];
  const float* lin_b     = (const float*)d_in[23];
  float* out = (float*)d_out;

  char* wsp = (char*)d_ws;
  size_t off = 0;
  auto carve = [&](size_t bytes) -> void* {
    off = (off + 255) & ~(size_t)255;
    void* p = wsp + off;
    off += bytes;
    return p;
  };
  float*  emb_f32   = (float*) carve((size_t)BB * DD * 4);   // masked in place later
  __bf16* flat_bf16 = (__bf16*)carve((size_t)BB * DD * 2);   // pre-mask, then overwritten masked
  float*  logits    = (float*) carve((size_t)BB * 48 * 4);
  __bf16* h0_bf16   = (__bf16*)carve((size_t)BB * H0 * 2);
  float*  h1_f32    = (float*) carve((size_t)BB * H1 * 4);
  float*  pp        = (float*) carve((size_t)BB * 4);
  __bf16* w0T       = (__bf16*)carve((size_t)H0 * DD * 2);   // pre-transposed [N][K]
  __bf16* w1T       = (__bf16*)carve((size_t)H1 * H0 * 2);   // pre-transposed [N][K]
  __bf16* ctrlT     = (__bf16*)carve((size_t)48 * DD * 2);
  float*  bnS = (float*)carve(FF * 4);
  float*  bnB = (float*)carve(FF * 4);
  float*  c0S = (float*)carve(H0 * 4);
  float*  c0B = (float*)carve(H0 * 4);
  float*  c1S = (float*)carve(H1 * 4);
  float*  c1B = (float*)carve(H1 * 4);

  prep_params_kernel<<<1, 1024, 0, stream>>>(bn_gamma, bn_beta, bn_mean, bn_var,
      mlp_b0, mlp_g0, mlp_be0, mlp_m0, mlp_v0,
      mlp_b1, mlp_g1, mlp_be1, mlp_m1, mlp_v1,
      bnS, bnB, c0S, c0B, c1S, c1B);
  transpose_to_bf16_kernel<<<(H0 * DD) / 256, 256, 0, stream>>>(mlp_w0, w0T, DD, H0);
  transpose_to_bf16_kernel<<<(H1 * H0) / 256, 256, 0, stream>>>(mlp_w1, w1T, H0, H1);
  build_ctrlT_kernel<<<(48 * DD) / 256, 256, 0, stream>>>(ctrl_w, ctrlT);

  gather_bn_kernel<<<(BB * DD) / 256, 256, 0, stream>>>(x, emb_table, bnS, bnB, emb_f32, flat_bf16);
  ctrl_gemm_kernel<<<BB / 16, 32, 0, stream>>>(flat_bf16, ctrlT, ctrl_b, logits);
  topk_mask_kernel<<<BB / 8, 256, 0, stream>>>(logits, emb_f32, flat_bf16);

  gemm_bn_relu_kernel<__bf16><<<dim3(H0 / 128, BB / 128), 256, 0, stream>>>(
      flat_bf16, w0T, c0S, c0B, h0_bf16, H0, DD);
  gemm_bn_relu_kernel<float><<<dim3(H1 / 128, BB / 128), 256, 0, stream>>>(
      h0_bf16, w1T, c1S, c1B, h1_f32, H1, H0);

  cross_kernel<<<BB / 8, 256, 0, stream>>>(emb_f32, cross_w, cross_b, lin_w, lin_b, pp);
  final_kernel<<<BB / 8, 256, 0, stream>>>(h1_f32, lin_w, pp, out);
}
